// MLA_30408368455966
// MI455X (gfx1250) — matrix operand order repaired and checked by reading, not compiled, not run on hardware
//
#include <hip/hip_runtime.h>
#include <hip/hip_bf16.h>

// ---------------------------------------------------------------------------
// MLA prefill for gfx1250 (CDNA5). All matmuls via v_wmma_f32_16x16x32_bf16.
// Round 2: software-pipelined GEMM (double-buffered fragments, 32x64 wave
// tile) and batched fragment loads in attention to eliminate the
// s_wait_loadcnt 0x0 full-serialization seen in round 1 asm.
// ---------------------------------------------------------------------------

typedef __bf16 bf16;
typedef __attribute__((ext_vector_type(8)))  bf16  v8bf;
typedef __attribute__((ext_vector_type(16))) bf16  v16bf;
typedef __attribute__((ext_vector_type(8)))  float v8f;

#define B_    2
#define S_    1024
#define D_    4096
#define H_    32
#define QL_   1536
#define KVL_  512
#define NOPE_ 128
#define ROPE_ 64
#define VD_   128
#define QKD_  192                      // NOPE + ROPE
#define KVOUT_ 256                     // NOPE + VD
#define FUSED_ 2112                    // QL + KVL + ROPE
#define ROWS_  (B_ * S_)               // 2048

__device__ __forceinline__ v8f wmma_bf16(v16bf a, v16bf b, v8f c) {
  return __builtin_amdgcn_wmma_f32_16x16x32_bf16(false, a, false, b,
                                                 (short)0, c, false, false);
}

__device__ __forceinline__ v8f zero_v8f() {
  v8f z;
#pragma unroll
  for (int i = 0; i < 8; ++i) z[i] = 0.0f;
  return z;
}

// A-fragment (16x32 bf16, row-major source). Contiguous-k layout used here (A and B alike):
// lane = m + 16*h ; element e -> k = e + 16*h
// -> two contiguous 8-bf16 (16B) loads per lane.
__device__ __forceinline__ v16bf load_a_frag(const bf16* __restrict__ A,
                                             int lda, int row0, int k0) {
  int lane = threadIdx.x & 31;
  int m = lane & 15, h = lane >> 4;
  const bf16* p = A + (size_t)(row0 + m) * lda + k0 + 16 * h;
  v8bf lo = *(const v8bf*)p;         // k = 16h .. 16h+7
  v8bf hi = *(const v8bf*)(p + 8);   // k = 16h+8 .. 16h+15
  v16bf r;
#pragma unroll
  for (int i = 0; i < 8; ++i) { r[i] = lo[i]; r[i + 8] = hi[i]; }
  return r;
}

// ---------------------------------------------------------------------------
// elementwise f32 -> bf16
__global__ void k_cvt_bf16(const float* __restrict__ x, bf16* __restrict__ y,
                           long n) {
  long i = (long)blockIdx.x * 256 + threadIdx.x;
  if (i < n) y[i] = (bf16)x[i];
}

// Pack weight [K,N] f32 row-major into WMMA B-fragment layout:
// idx = (((kt*(N/16)+nt)*2 + h)*16 + n)*16 + e  holds  B[kt*32+16h+e][nt*16+n]
// -> each lane's B-fragment is one contiguous 32B v16bf load.
__global__ void k_pack_w(const float* __restrict__ w, bf16* __restrict__ wp,
                         int K, int N) {
  long i = (long)blockIdx.x * 256 + threadIdx.x;
  long total = (long)K * N;
  if (i >= total) return;
  int e = (int)(i & 15);
  int n = (int)((i >> 4) & 15);
  int h = (int)((i >> 8) & 1);
  long rest = i >> 9;
  int ntiles = N >> 4;
  int nt = (int)(rest % ntiles);
  int kt = (int)(rest / ntiles);
  int k = kt * 32 + 16 * h + e;
  wp[i] = (bf16)w[(size_t)k * N + nt * 16 + n];
}

// RMSNorm over a column slice of `fused`, writing bf16.
__global__ void k_rmsnorm(const float* __restrict__ fused,
                          const float* __restrict__ w, bf16* __restrict__ out,
                          int cols, int src_ld, int src_off) {
  int row = blockIdx.x, tid = threadIdx.x;
  const float* src = fused + (size_t)row * src_ld + src_off;
  __shared__ float red[256];
  float ss = 0.0f;
  for (int c = tid; c < cols; c += 256) { float v = src[c]; ss += v * v; }
  red[tid] = ss;
  __syncthreads();
  for (int s = 128; s > 0; s >>= 1) {
    if (tid < s) red[tid] += red[tid + s];
    __syncthreads();
  }
  float inv = rsqrtf(red[0] / (float)cols + 1e-6f);
  for (int c = tid; c < cols; c += 256)
    out[(size_t)row * cols + c] = (bf16)(src[c] * inv * w[c]);
}

// GEMM: C[M,N] f32 = A[M,K] bf16 row-major x Bp (packed) bf16.
// Block = 256 threads = 8 waves; block tile 256 rows x 64 cols.
// Each wave: 32 rows x 64 cols = 8 WMMAs per 32-K step (A reuse x4, B x2).
// Software pipelined: next K-step fragments are loaded before the current
// step's WMMAs so load latency hides behind 8 matrix ops.
__global__ void __launch_bounds__(256)
k_gemm(const bf16* __restrict__ A, const bf16* __restrict__ Bp,
       float* __restrict__ C, int M, int N, int K) {
  int lane = threadIdx.x & 31;
  int wave = threadIdx.x >> 5;
  int n = lane & 15, h = lane >> 4;
  int row0 = blockIdx.y * 256 + wave * 32;
  int col0 = blockIdx.x * 64;
  int ntiles = N >> 4;
  // element offset: kt*ntiles*512 + nt*512 + h*256 + n*16
  const bf16* bbase = Bp + (size_t)(col0 >> 4) * 512 + h * 256 + n * 16;
  size_t bkt = (size_t)ntiles * 512;

  v8f acc[8];
#pragma unroll
  for (int t = 0; t < 8; ++t) acc[t] = zero_v8f();

  v16bf a0 = load_a_frag(A, K, row0, 0);
  v16bf a1 = load_a_frag(A, K, row0 + 16, 0);
  v16bf b0 = *(const v16bf*)(bbase);
  v16bf b1 = *(const v16bf*)(bbase + 512);
  v16bf b2 = *(const v16bf*)(bbase + 1024);
  v16bf b3 = *(const v16bf*)(bbase + 1536);

  for (int k0 = 0; k0 + 32 < K; k0 += 32) {
    int k1 = k0 + 32;
    // speculative prefetch two steps ahead (safe out-of-range per ISA)
    __builtin_prefetch(A + (size_t)(row0 + (lane & 15)) * K + k1 + 32, 0, 1);
    v16bf na0 = load_a_frag(A, K, row0, k1);
    v16bf na1 = load_a_frag(A, K, row0 + 16, k1);
    const bf16* bb = bbase + (size_t)(k1 >> 5) * bkt;
    v16bf nb0 = *(const v16bf*)(bb);
    v16bf nb1 = *(const v16bf*)(bb + 512);
    v16bf nb2 = *(const v16bf*)(bb + 1024);
    v16bf nb3 = *(const v16bf*)(bb + 1536);
    acc[0] = wmma_bf16(a0, b0, acc[0]);
    acc[4] = wmma_bf16(a1, b0, acc[4]);
    acc[1] = wmma_bf16(a0, b1, acc[1]);
    acc[5] = wmma_bf16(a1, b1, acc[5]);
    acc[2] = wmma_bf16(a0, b2, acc[2]);
    acc[6] = wmma_bf16(a1, b2, acc[6]);
    acc[3] = wmma_bf16(a0, b3, acc[3]);
    acc[7] = wmma_bf16(a1, b3, acc[7]);
    a0 = na0; a1 = na1;
    b0 = nb0; b1 = nb1; b2 = nb2; b3 = nb3;
  }
  // epilogue K-step
  acc[0] = wmma_bf16(a0, b0, acc[0]);
  acc[4] = wmma_bf16(a1, b0, acc[4]);
  acc[1] = wmma_bf16(a0, b1, acc[1]);
  acc[5] = wmma_bf16(a1, b1, acc[5]);
  acc[2] = wmma_bf16(a0, b2, acc[2]);
  acc[6] = wmma_bf16(a1, b2, acc[6]);
  acc[3] = wmma_bf16(a0, b3, acc[3]);
  acc[7] = wmma_bf16(a1, b3, acc[7]);

#pragma unroll
  for (int at = 0; at < 2; ++at)
#pragma unroll
    for (int t = 0; t < 4; ++t)
#pragma unroll
      for (int r = 0; r < 8; ++r) {
        int row = row0 + 16 * at + r + 8 * h;
        int col = col0 + 16 * t + n;
        C[(size_t)row * N + col] = acc[at * 4 + t][r];
      }
}

// RoPE helper: half-rotation tables computed on the fly.
__device__ __forceinline__ void rope_cs(int pos, int i32, float* sn, float* cs) {
  // inv_freq = 10000^(-i/32) = 2^(-i/32 * log2(10000))
  float inv = exp2f(-13.287712379549449f * (float)i32 * (1.0f / 32.0f));
  __sincosf((float)pos * inv, sn, cs);
}

// q f32 [rows, H*192] -> qf bf16 [b,h,s,192] with RoPE on last 64 and
// softmax scale folded in.
__global__ void k_pack_q(const float* __restrict__ q, bf16* __restrict__ qf) {
  long i = (long)blockIdx.x * 256 + threadIdx.x;
  const long total = (long)ROWS_ * H_ * QKD_;
  if (i >= total) return;
  int d = (int)(i % QKD_);
  long t = i / QKD_;
  int hh = (int)(t % H_);
  long row = t / H_;          // b*S + s
  int s = (int)(row % S_);
  int b = (int)(row / S_);
  const float* src = q + (size_t)row * (H_ * QKD_) + hh * QKD_;
  const float scale = 0.07216878364870323f;  // 1/sqrt(192)
  float val;
  if (d < NOPE_) {
    val = src[d];
  } else {
    int ri = d - NOPE_;
    int i32 = ri & 31;
    float sn, cs;
    rope_cs(s, i32, &sn, &cs);
    float x1 = src[NOPE_ + i32];
    float x2 = src[NOPE_ + 32 + i32];
    val = (ri < 32) ? (x1 * cs - x2 * sn) : (x1 * sn + x2 * cs);
  }
  qf[(((size_t)b * H_ + hh) * S_ + s) * QKD_ + d] = (bf16)(val * scale);
}

// kmat bf16 [b,h,s,192] = concat(k_nope[b,s,h,:], rope(k_pe[b,s,:]) bcast)
__global__ void k_pack_k(const float* __restrict__ kv,
                         const float* __restrict__ fused,
                         bf16* __restrict__ kmat) {
  long i = (long)blockIdx.x * 256 + threadIdx.x;
  const long total = (long)ROWS_ * H_ * QKD_;
  if (i >= total) return;
  int d = (int)(i % QKD_);
  long t = i / QKD_;
  int hh = (int)(t % H_);
  long row = t / H_;
  int s = (int)(row % S_);
  int b = (int)(row / S_);
  float val;
  if (d < NOPE_) {
    val = kv[(size_t)row * (H_ * KVOUT_) + hh * KVOUT_ + d];
  } else {
    const float* kp = fused + (size_t)row * FUSED_ + QL_ + KVL_;
    int ri = d - NOPE_;
    int i32 = ri & 31;
    float sn, cs;
    rope_cs(s, i32, &sn, &cs);
    float x1 = kp[i32];
    float x2 = kp[32 + i32];
    val = (ri < 32) ? (x1 * cs - x2 * sn) : (x1 * sn + x2 * cs);
  }
  kmat[(((size_t)b * H_ + hh) * S_ + s) * QKD_ + d] = (bf16)val;
}

// vT bf16 [b,h,d,s] = kv[b,s,h,128+d]  (so V^T A-fragments are contiguous)
__global__ void k_pack_vt(const float* __restrict__ kv, bf16* __restrict__ vT) {
  long i = (long)blockIdx.x * 256 + threadIdx.x;
  const long total = (long)B_ * H_ * VD_ * S_;
  if (i >= total) return;
  int s = (int)(i % S_);
  long t = i / S_;
  int d = (int)(t % VD_);
  t /= VD_;
  int hh = (int)(t % H_);
  int b = (int)(t / H_);
  vT[i] = (bf16)kv[((size_t)(b * S_ + s)) * (H_ * KVOUT_) + hh * KVOUT_ +
                   NOPE_ + d];
}

// Causal flash attention, transposed-tile formulation.
// One wave per (b, h, 16-query block). S^T = K.Q^T (12 WMMA / 32 keys),
// ctx^T = V^T.P^T (8 WMMA / 32 keys). Softmax stats are per-lane (query ==
// lane column): one shfl_xor(16) per reduction. All K-fragments for a key
// block are loaded before the score WMMAs, and V-fragments are loaded before
// the softmax VALU work, so load latency hides behind matrix/exp math.
__global__ void __launch_bounds__(128)
k_attn(const bf16* __restrict__ qf, const bf16* __restrict__ kmat,
       const bf16* __restrict__ vT, bf16* __restrict__ ctx) {
  int lane = threadIdx.x & 31;
  int wid = blockIdx.x * 4 + (threadIdx.x >> 5);
  int qb = wid & 63;          // S/16 query blocks
  int bh = wid >> 6;          // b*H + h
  int m0 = qb * 16;
  int n = lane & 15, hh = lane >> 4;

  const bf16* qf_b = qf + (size_t)bh * S_ * QKD_;
  const bf16* k_b  = kmat + (size_t)bh * S_ * QKD_;
  const bf16* v_b  = vT + (size_t)bh * VD_ * S_;

  // Q^T B-fragments: lane n = query, element e -> feature 32t + 16hh + e
  v16bf qfrag[6];
#pragma unroll
  for (int t = 0; t < 6; ++t)
    qfrag[t] = *(const v16bf*)(qf_b + (size_t)(m0 + n) * QKD_ + 32 * t + 16 * hh);

  v8f acc[8];
#pragma unroll
  for (int dt = 0; dt < 8; ++dt) acc[dt] = zero_v8f();

  const float L2E = 1.4426950408889634f;
  float mx = -1e30f, l = 0.0f;
  int q = m0 + n;

  for (int k0 = 0; k0 < m0 + 16; k0 += 32) {
    // batch-load all 12 K fragments, then run 12 WMMAs (partial load waits)
    v16bf ka[6], kb[6];
#pragma unroll
    for (int t = 0; t < 6; ++t) {
      ka[t] = load_a_frag(k_b, QKD_, k0, 32 * t);
      kb[t] = load_a_frag(k_b, QKD_, k0 + 16, 32 * t);
    }
    v8f s1 = zero_v8f(), s2 = zero_v8f();
#pragma unroll
    for (int t = 0; t < 6; ++t) {
      s1 = wmma_bf16(ka[t], qfrag[t], s1);
      s2 = wmma_bf16(kb[t], qfrag[t], s2);
    }
    // issue V^T fragment loads now; softmax VALU below hides their latency
    v16bf va[8];
#pragma unroll
    for (int dt = 0; dt < 8; ++dt)
      va[dt] = load_a_frag(v_b, S_, dt * 16, k0);

    // causal mask + block max (branchless; EXEC stays all-ones)
    float p1[8], p2[8];
    float bm = -1e30f;
#pragma unroll
    for (int r = 0; r < 8; ++r) {
      int key = k0 + r + 8 * hh;
      float a = (key <= q) ? s1[r] : -1e30f;
      float c = (key + 16 <= q) ? s2[r] : -1e30f;
      p1[r] = a; p2[r] = c;
      bm = fmaxf(bm, fmaxf(a, c));
    }
    bm = fmaxf(bm, __shfl_xor(bm, 16));
    float nm = fmaxf(mx, bm);
    float corr = exp2f((mx - nm) * L2E);
    mx = nm;
    float rs = 0.0f;
#pragma unroll
    for (int r = 0; r < 8; ++r) {
      p1[r] = exp2f((p1[r] - nm) * L2E);
      p2[r] = exp2f((p2[r] - nm) * L2E);
      rs += p1[r] + p2[r];
    }
    rs += __shfl_xor(rs, 16);
    l = l * corr + rs;
#pragma unroll
    for (int dt = 0; dt < 8; ++dt)
#pragma unroll
      for (int r = 0; r < 8; ++r) acc[dt][r] = acc[dt][r] * corr;

    // P^T B-fragment: C-layout -> B-layout needs one cross-half exchange.
    v16bf pb;
#pragma unroll
    for (int r = 0; r < 8; ++r) {
      float o1 = __shfl_xor(p1[r], 16);
      float o2 = __shfl_xor(p2[r], 16);
      pb[r]     = (bf16)(hh ? o2 : p1[r]);
      pb[r + 8] = (bf16)(hh ? p2[r] : o1);
    }
#pragma unroll
    for (int dt = 0; dt < 8; ++dt)
      acc[dt] = wmma_bf16(va[dt], pb, acc[dt]);
  }

  float invl = 1.0f / l;
  int b = bh >> 5;
  int h = bh & 31;
  size_t orow = ((size_t)(b * S_ + m0 + n)) * (H_ * VD_) + (size_t)h * VD_;
#pragma unroll
  for (int dt = 0; dt < 8; ++dt)
#pragma unroll
    for (int r = 0; r < 8; ++r)
      ctx[orow + dt * 16 + r + 8 * hh] = (bf16)(acc[dt][r] * invl);
}

// ---------------------------------------------------------------------------
extern "C" void kernel_launch(void* const* d_in, const int* in_sizes, int n_in,
                              void* d_out, int out_size, void* d_ws,
                              size_t ws_size, hipStream_t stream) {
  (void)in_sizes; (void)n_in; (void)out_size; (void)ws_size;
  const float* hidden    = (const float*)d_in[0];
  const float* w_fused_a = (const float*)d_in[1];
  const float* q_ln      = (const float*)d_in[2];
  const float* kv_ln     = (const float*)d_in[3];
  const float* w_q_b     = (const float*)d_in[4];
  const float* w_kv_b    = (const float*)d_in[5];
  const float* w_o       = (const float*)d_in[6];
  float* out = (float*)d_out;

  char* ws = (char*)d_ws;
  size_t off = 0;
  auto alloc = [&](size_t bytes) -> void* {
    void* p = ws + off;
    off += (bytes + 255) & ~(size_t)255;
    return p;
  };

  // packW is reused for all four weights (max = w_o: 4096x4096 bf16).
  bf16* packW = (bf16*)alloc((size_t)D_ * D_ * sizeof(bf16));
  float* fused = (float*)alloc((size_t)ROWS_ * FUSED_ * sizeof(float));
  bf16* qn  = (bf16*)alloc((size_t)ROWS_ * QL_ * sizeof(bf16));
  bf16* kvn = (bf16*)alloc((size_t)ROWS_ * KVL_ * sizeof(bf16));
  bf16* qfb  = (bf16*)alloc((size_t)B_ * H_ * S_ * QKD_ * sizeof(bf16));
  bf16* kmat = (bf16*)alloc((size_t)B_ * H_ * S_ * QKD_ * sizeof(bf16));
  bf16* vT   = (bf16*)alloc((size_t)B_ * H_ * VD_ * S_ * sizeof(bf16));
  bf16* ctx  = (bf16*)alloc((size_t)ROWS_ * (H_ * VD_) * sizeof(bf16));
  // big scratch: hs_bf (16MB) -> qbuf (50MB) -> kvbuf (67MB), lifetimes disjoint
  void* big = alloc((size_t)ROWS_ * (H_ * KVOUT_) * sizeof(float));
  bf16* hs_bf  = (bf16*)big;
  float* qbuf  = (float*)big;
  float* kvbuf = (float*)big;

  // 1) hidden -> bf16
  long nh = (long)ROWS_ * D_;
  k_cvt_bf16<<<(nh + 255) / 256, 256, 0, stream>>>(hidden, hs_bf, nh);

  // 2) fused = hidden @ w_fused_a   [2048 x 2112]
  long nw = (long)D_ * FUSED_;
  k_pack_w<<<(nw + 255) / 256, 256, 0, stream>>>(w_fused_a, packW, D_, FUSED_);
  {
    dim3 g(FUSED_ / 64, ROWS_ / 256);
    k_gemm<<<g, 256, 0, stream>>>(hs_bf, packW, fused, ROWS_, FUSED_, D_);
  }

  // 3) RMSNorms -> bf16 latents
  k_rmsnorm<<<ROWS_, 256, 0, stream>>>(fused, q_ln, qn, QL_, FUSED_, 0);
  k_rmsnorm<<<ROWS_, 256, 0, stream>>>(fused, kv_ln, kvn, KVL_, FUSED_, QL_);

  // 4) q = qn @ w_q_b  [2048 x 6144] -> qbuf (f32), then RoPE+scale pack
  nw = (long)QL_ * (H_ * QKD_);
  k_pack_w<<<(nw + 255) / 256, 256, 0, stream>>>(w_q_b, packW, QL_, H_ * QKD_);
  {
    dim3 g((H_ * QKD_) / 64, ROWS_ / 256);
    k_gemm<<<g, 256, 0, stream>>>(qn, packW, qbuf, ROWS_, H_ * QKD_, QL_);
  }
  long nq = (long)ROWS_ * H_ * QKD_;
  k_pack_q<<<(nq + 255) / 256, 256, 0, stream>>>(qbuf, qfb);

  // 5) kv = kvn @ w_kv_b  [2048 x 8192] -> kvbuf (f32), then K/V packs
  nw = (long)KVL_ * (H_ * KVOUT_);
  k_pack_w<<<(nw + 255) / 256, 256, 0, stream>>>(w_kv_b, packW, KVL_,
                                                 H_ * KVOUT_);
  {
    dim3 g((H_ * KVOUT_) / 64, ROWS_ / 256);
    k_gemm<<<g, 256, 0, stream>>>(kvn, packW, kvbuf, ROWS_, H_ * KVOUT_, KVL_);
  }
  k_pack_k<<<(nq + 255) / 256, 256, 0, stream>>>(kvbuf, fused, kmat);
  long nv = (long)B_ * H_ * VD_ * S_;
  k_pack_vt<<<(nv + 255) / 256, 256, 0, stream>>>(kvbuf, vT);

  // 6) attention: 4096 waves, 4 waves/block
  k_attn<<<(B_ * H_ * (S_ / 16)) / 4, 128, 0, stream>>>(qfb, kmat, vT, ctx);

  // 7) out = ctx @ w_o  [2048 x 4096]
  nw = (long)(H_ * VD_) * D_;
  k_pack_w<<<(nw + 255) / 256, 256, 0, stream>>>(w_o, packW, H_ * VD_, D_);
  {
    dim3 g(D_ / 64, ROWS_ / 256);
    k_gemm<<<g, 256, 0, stream>>>(ctx, packW, out, ROWS_, D_, H_ * VD_);
  }
}
